// CausalSelfAttention_3161095930006
// MI455X (gfx1250) — matrix operand order repaired and checked by reading, not compiled, not run on hardware
//
#include <hip/hip_runtime.h>
#include <math.h>

typedef __attribute__((ext_vector_type(16))) __bf16 v16bf;
typedef __attribute__((ext_vector_type(8)))  float  v8f;
typedef __attribute__((ext_vector_type(4)))  unsigned int v4u;
typedef __attribute__((ext_vector_type(8)))  unsigned int v8u;
typedef unsigned short u16;

#define N_EMBD   1024
#define N_HEAD   16
#define HEAD_DIM 64
#define T_SEQ    2048
#define BATCH    4
#define M_ROWS   (BATCH * T_SEQ)   // 8192

static __device__ __forceinline__ v8f wmma_bf16(v16bf a, v16bf b, v8f c) {
  return __builtin_amdgcn_wmma_f32_16x16x32_bf16(false, a, false, b, (short)0, c, false, false);
}

static __device__ __forceinline__ u16 f2bf(float f) {
  return __builtin_bit_cast(u16, (__bf16)f);
}

// Build a 16-half fragment from two 16-byte chunks (elements 0-7, 8-15).
static __device__ __forceinline__ v16bf frag2(const u16* p0, const u16* p1) {
  v4u lo = *(const v4u*)p0;
  v4u hi = *(const v4u*)p1;
  v8u t = {lo[0], lo[1], lo[2], lo[3], hi[0], hi[1], hi[2], hi[3]};
  return __builtin_bit_cast(v16bf, t);
}

// ---------------------------------------------------------------------------
// f32 -> bf16 elementwise (x): 4 elems / thread
// ---------------------------------------------------------------------------
__global__ __launch_bounds__(256)
void cvt_x_kernel(const float* __restrict__ src, u16* __restrict__ dst) {
  size_t i = ((size_t)blockIdx.x * 256 + threadIdx.x) * 4;
  float4 f = *(const float4*)(src + i);
  unsigned int lo = (unsigned int)f2bf(f.x) | ((unsigned int)f2bf(f.y) << 16);
  unsigned int hi = (unsigned int)f2bf(f.z) | ((unsigned int)f2bf(f.w) << 16);
  uint2 p; p.x = lo; p.y = hi;
  *(uint2*)(dst + i) = p;
}

// ---------------------------------------------------------------------------
// W [K x N] f32  ->  Wt [N x K] bf16   (coalesced writes; reads are L2-hot)
// ---------------------------------------------------------------------------
__global__ __launch_bounds__(256)
void cvt_w_t_kernel(const float* W0, const float* W1, const float* W2, const float* W3,
                    u16* T0, u16* T1, u16* T2, u16* T3) {
  const int z = blockIdx.z;
  const float* W = (z == 0) ? W0 : (z == 1) ? W1 : (z == 2) ? W2 : W3;
  u16* T = (z == 0) ? T0 : (z == 1) ? T1 : (z == 2) ? T2 : T3;
  int kk = blockIdx.x * 256 + threadIdx.x;
  int n  = blockIdx.y;
  T[(size_t)n * N_EMBD + kk] = f2bf(W[(size_t)kk * N_EMBD + n]);
}

// ---------------------------------------------------------------------------
// GEMM (bf16 in, f32 acc): [M x 1024] @ Wt^T + bias.
// Block = 256 thr (8 waves); block tile 64(M) x 256(N); wave tile 32x64
// (8 WMMA accumulators); K-step 32; double-buffered LDS; fragments preloaded
// into distinct registers so DS waits stagger under the WMMA chain.
// ---------------------------------------------------------------------------
#define LDA 40   // halves; 80B row stride: 16B-aligned, conflict-free

__global__ __launch_bounds__(256)
void gemm_bf16(const u16* __restrict__ X,
               const u16* __restrict__ Wt0, const u16* __restrict__ Wt1,
               const u16* __restrict__ Wt2,
               const float* __restrict__ b0, const float* __restrict__ b1,
               const float* __restrict__ b2,
               float* __restrict__ F1, float* __restrict__ F2,
               float* __restrict__ Fp,
               u16* __restrict__ H0, u16* __restrict__ H1, u16* __restrict__ H2,
               int mode)
{
  __shared__ __attribute__((aligned(16))) u16 As[2][64 * LDA];
  __shared__ __attribute__((aligned(16))) u16 Bs[2][256 * LDA];

  const int z = blockIdx.z;
  const u16*   Wt   = (z == 0) ? Wt0 : (z == 1) ? Wt1 : Wt2;
  const float* bias = (z == 0) ? b0  : (z == 1) ? b1  : b2;

  const int tid  = threadIdx.x;
  const int lane = tid & 31;
  const int wave = tid >> 5;
  const int wm   = wave >> 2;       // 0..1
  const int wn   = wave & 3;        // 0..3
  const int row_blk = blockIdx.y * 64;
  const int col_blk = blockIdx.x * 256;

  const int lr  = lane & 15;
  const int hi  = lane >> 4;
  const int kbA = hi * 8;
  const int kbB = hi * 8;

  v8f acc[2][4];
#pragma unroll
  for (int a = 0; a < 2; ++a)
#pragma unroll
    for (int b = 0; b < 4; ++b) acc[a][b] = (v8f){};

  // cooperative tile load: A 64x32 halves (1 x b128/thread),
  // B 256 cols x 32 halves (4 x b128/thread, one column per thread)
  const int lar = tid >> 2, lak = (tid & 3) * 8;
  auto load_tiles = [&](int k0, int buf) {
    *(v4u*)&As[buf][lar * LDA + lak] =
        *(const v4u*)&X[(size_t)(row_blk + lar) * N_EMBD + k0 + lak];
    const u16* srcB = &Wt[(size_t)(col_blk + tid) * N_EMBD + k0];
    u16* dstB = &Bs[buf][tid * LDA];
    *(v4u*)(dstB)      = *(const v4u*)(srcB);
    *(v4u*)(dstB + 8)  = *(const v4u*)(srcB + 8);
    *(v4u*)(dstB + 16) = *(const v4u*)(srcB + 16);
    *(v4u*)(dstB + 24) = *(const v4u*)(srcB + 24);
  };

  const int nsteps = N_EMBD / 32;
  load_tiles(0, 0);
  int cur = 0;

  for (int s = 0; s < nsteps; ++s) {
    __syncthreads();
    if (s + 1 < nsteps) {
      load_tiles((s + 1) * 32, cur ^ 1);
      if (s + 2 < nsteps)  // gfx1250 global_prefetch_b8 for the tile after
        __builtin_prefetch(&X[(size_t)(row_blk + lar) * N_EMBD + (s + 2) * 32 + lak]);
    }

    // preload ALL fragments first (distinct registers -> staggered dscnt waits)
    const u16* a0p = &As[cur][(wm * 32 + lr) * LDA + kbA];
    const u16* a1p = a0p + 16 * LDA;
    v16bf a0 = frag2(a0p, a0p + 16);
    v16bf a1 = frag2(a1p, a1p + 16);

    v16bf bbf[4];
#pragma unroll
    for (int j = 0; j < 4; ++j) {
      const u16* bp = &Bs[cur][(wn * 64 + j * 16 + lr) * LDA + kbB];
      bbf[j] = frag2(bp, bp + 16);
    }

#pragma unroll
    for (int j = 0; j < 4; ++j) {
      acc[0][j] = wmma_bf16(a0, bbf[j], acc[0][j]);
      acc[1][j] = wmma_bf16(a1, bbf[j], acc[1][j]);
    }
    cur ^= 1;
  }

  // epilogue
  auto store = [&](int m, int n, float val) {
    if (mode == 0) {
      Fp[(size_t)m * N_EMBD + n] = val;
    } else {
      int bb = m >> 11, t = m & (T_SEQ - 1);
      int h  = n >> 6,  d = n & (HEAD_DIM - 1);
      size_t idx = (((size_t)(bb * N_HEAD + h)) * T_SEQ + t) * HEAD_DIM + d;
      u16 hv = f2bf(val);
      if (z == 0) {
        H0[idx] = hv;                                  // q bf16
      } else if (z == 1) {
        F1[idx] = val; H1[idx] = hv;                   // k f32 + bf16
      } else {
        F2[idx] = val;                                 // v f32
        H2[(((size_t)(bb * N_HEAD + h)) * HEAD_DIM + d) * T_SEQ + t] = hv; // v^T bf16
      }
    }
  };

#pragma unroll
  for (int jm = 0; jm < 2; ++jm) {
    const int mrow = row_blk + wm * 32 + jm * 16 + hi * 8;
#pragma unroll
    for (int jn = 0; jn < 4; ++jn) {
      const int nc = col_blk + wn * 64 + jn * 16 + lr;
      const float bval = bias[nc];
#pragma unroll
      for (int i = 0; i < 8; ++i)
        store(mrow + i, nc, acc[jm][jn][i] + bval);
    }
  }
}

// ---------------------------------------------------------------------------
// Flash attention, bf16 operands, f32 online softmax.
// One wave = 16 query rows; streams 32-key tiles: 4 WMMA (QK^T) +
// softmax (width-16 shuffle reductions) + 4 WMMA (P V).
// ---------------------------------------------------------------------------
#define LDP 40

__global__ __launch_bounds__(256)
void attn_kernel(const u16* __restrict__ Q, const u16* __restrict__ Kb,
                 const u16* __restrict__ Vt, u16* __restrict__ Y)
{
  __shared__ __attribute__((aligned(16))) u16 plds[8][16 * LDP];

  const int tid  = threadIdx.x;
  const int lane = tid & 31;
  const int wave = tid >> 5;
  const int lr   = lane & 15;
  const int hi   = lane >> 4;
  const int kbA  = hi * 8;
  const int kbB  = hi * 8;

  const int h  = blockIdx.y;
  const int b  = blockIdx.z;
  const int r0 = blockIdx.x * 128 + wave * 16;

  const size_t bh  = ((size_t)b * N_HEAD + h) * (size_t)T_SEQ * HEAD_DIM;
  const u16* q  = Q  + bh;
  const u16* k  = Kb + bh;
  const u16* vt = Vt + bh;    // [D][T] layout

  // Q fragments d=[0,32), [32,64), loaded once
  const u16* qr = q + (size_t)(r0 + lr) * HEAD_DIM;
  v16bf aq0 = frag2(qr + kbA,      qr + kbA + 16);
  v16bf aq1 = frag2(qr + 32 + kbA, qr + 32 + kbA + 16);

  v8f o0 = {}, o1 = {}, o2 = {}, o3 = {};
  float m_i[8], l_i[8];
#pragma unroll
  for (int i = 0; i < 8; ++i) { m_i[i] = -__builtin_inff(); l_i[i] = 0.f; }

  const float scale = 0.125f;
  const int   row   = r0 + hi * 8;

  for (int s0 = 0; s0 <= r0 + 15; s0 += 32) {
    // ---- preload K fragments for both 16-column tiles, then 4 WMMA
    const u16* kc0 = k + (size_t)(s0 + lr) * HEAD_DIM;
    const u16* kc1 = k + (size_t)(s0 + 16 + lr) * HEAD_DIM;
    v16bf bk00 = frag2(kc0 + kbB,      kc0 + kbB + 16);
    v16bf bk01 = frag2(kc0 + 32 + kbB, kc0 + 32 + kbB + 16);
    v16bf bk10 = frag2(kc1 + kbB,      kc1 + kbB + 16);
    v16bf bk11 = frag2(kc1 + 32 + kbB, kc1 + 32 + kbB + 16);
    __builtin_prefetch(k + (size_t)(s0 + 32 + lr) * HEAD_DIM);

    v8f s_c0 = {}, s_c1 = {};
    s_c0 = wmma_bf16(aq0, bk00, s_c0);
    s_c0 = wmma_bf16(aq1, bk01, s_c0);
    s_c1 = wmma_bf16(aq0, bk10, s_c1);
    s_c1 = wmma_bf16(aq1, bk11, s_c1);

    // ---- online softmax
#pragma unroll
    for (int i = 0; i < 8; ++i) {
      const int rr = row + i;
      float v0 = (s0 + lr      <= rr) ? s_c0[i] * scale : -__builtin_inff();
      float v1 = (s0 + 16 + lr <= rr) ? s_c1[i] * scale : -__builtin_inff();
      float rm = fmaxf(v0, v1);
#pragma unroll
      for (int off = 8; off > 0; off >>= 1)
        rm = fmaxf(rm, __shfl_xor(rm, off, 16));
      const float mn = fmaxf(m_i[i], rm);
      const float f  = __expf(m_i[i] - mn);
      const float p0 = __expf(v0 - mn);
      const float p1 = __expf(v1 - mn);
      float rs = p0 + p1;
#pragma unroll
      for (int off = 8; off > 0; off >>= 1)
        rs += __shfl_xor(rs, off, 16);
      l_i[i] = l_i[i] * f + rs;
      m_i[i] = mn;
      s_c0[i] = p0; s_c1[i] = p1;
      o0[i] *= f; o1[i] *= f; o2[i] *= f; o3[i] *= f;
    }

    // ---- P: C-layout -> per-wave LDS (bf16) -> A-layout
    u16* pw = plds[wave];
#pragma unroll
    for (int i = 0; i < 8; ++i) {
      pw[(hi * 8 + i) * LDP + lr]      = f2bf(s_c0[i]);
      pw[(hi * 8 + i) * LDP + 16 + lr] = f2bf(s_c1[i]);
    }
    asm volatile("" ::: "memory");
    const u16* pr = &pw[lr * LDP + kbA];
    v16bf ap = frag2(pr, pr + 16);
    asm volatile("" ::: "memory");

    // ---- preload V^T fragments (distinct regs), then 4 WMMA accumulate
    const u16* vr = vt + (size_t)lr * T_SEQ + s0 + kbB;
    v16bf bv0 = frag2(vr,                    vr + 16);
    v16bf bv1 = frag2(vr + 16 * T_SEQ,       vr + 16 * T_SEQ + 16);
    v16bf bv2 = frag2(vr + 32 * T_SEQ,       vr + 32 * T_SEQ + 16);
    v16bf bv3 = frag2(vr + 48 * T_SEQ,       vr + 48 * T_SEQ + 16);
    o0 = wmma_bf16(ap, bv0, o0);
    o1 = wmma_bf16(ap, bv1, o1);
    o2 = wmma_bf16(ap, bv2, o2);
    o3 = wmma_bf16(ap, bv3, o3);
  }

  // ---- normalize; store bf16 y_att [B,T,C] (input to projection GEMM)
  u16* yb = Y + (size_t)b * T_SEQ * N_EMBD + (size_t)h * HEAD_DIM;
#pragma unroll
  for (int i = 0; i < 8; ++i) {
    const float inv = 1.0f / l_i[i];
    u16* yr = yb + (size_t)(row + i) * N_EMBD;
    yr[lr]      = f2bf(o0[i] * inv);
    yr[16 + lr] = f2bf(o1[i] * inv);
    yr[32 + lr] = f2bf(o2[i] * inv);
    yr[48 + lr] = f2bf(o3[i] * inv);
  }
}

// ---------------------------------------------------------------------------
extern "C" void kernel_launch(void* const* d_in, const int* in_sizes, int n_in,
                              void* d_out, int out_size, void* d_ws, size_t ws_size,
                              hipStream_t stream) {
  const float* x  = (const float*)d_in[0];
  const float* Wq = (const float*)d_in[1];
  const float* bq = (const float*)d_in[2];
  const float* Wk = (const float*)d_in[3];
  const float* bk = (const float*)d_in[4];
  const float* Wv = (const float*)d_in[5];
  const float* bv = (const float*)d_in[6];
  const float* Wp = (const float*)d_in[7];
  const float* bp = (const float*)d_in[8];

  float* out = (float*)d_out;
  const size_t Y_SZ  = (size_t)BATCH * T_SEQ * N_EMBD;             // 8388608
  const size_t KV_SZ = (size_t)BATCH * N_HEAD * T_SEQ * HEAD_DIM;  // 8388608
  const size_t W_SZ  = (size_t)N_EMBD * N_EMBD;                    // 1048576
  float* y_out = out;
  float* k_out = out + Y_SZ;          // present[0]
  float* v_out = out + Y_SZ + KV_SZ;  // present[1]

  u16* ws   = (u16*)d_ws;
  u16* xb   = ws;                 // x bf16            [M,K]
  u16* wtq  = xb  + Y_SZ;         // Wq^T bf16         [N,K]
  u16* wtk  = wtq + W_SZ;
  u16* wtv  = wtk + W_SZ;
  u16* wtp  = wtv + W_SZ;
  u16* qb   = wtp + W_SZ;         // q  bf16 [B,H,T,D]
  u16* kb   = qb  + KV_SZ;        // k  bf16 [B,H,T,D]
  u16* vtb  = kb  + KV_SZ;        // v^T bf16 [B,H,D,T]
  u16* yatb = vtb + KV_SZ;        // y_att bf16 [B,T,C]

  dim3 blk(256);

  // 1) precision conversion (once per call)
  cvt_x_kernel<<<dim3(Y_SZ / (256 * 4)), blk, 0, stream>>>(x, xb);
  cvt_w_t_kernel<<<dim3(N_EMBD / 256, N_EMBD, 4), blk, 0, stream>>>(
      Wq, Wk, Wv, Wp, wtq, wtk, wtv, wtp);

  // 2) fused Q/K/V projections (k,v f32 straight into `present`)
  dim3 g1(N_EMBD / 256, M_ROWS / 64, 3);
  gemm_bf16<<<g1, blk, 0, stream>>>(xb, wtq, wtk, wtv, bq, bk, bv,
                                    k_out, v_out, nullptr,
                                    qb, kb, vtb, /*mode=*/1);

  // 3) causal flash attention
  dim3 g2(T_SEQ / 128, N_HEAD, BATCH);
  attn_kernel<<<g2, blk, 0, stream>>>(qb, kb, vtb, yatb);

  // 4) output projection
  dim3 g3(N_EMBD / 256, M_ROWS / 64, 1);
  gemm_bf16<<<g3, blk, 0, stream>>>(yatb, wtp, wtp, wtp, bp, bp, bp,
                                    nullptr, nullptr, y_out,
                                    nullptr, nullptr, nullptr, /*mode=*/0);
}
